// NAT_46643344835225
// MI455X (gfx1250) — matrix operand order repaired and checked by reading, not compiled, not run on hardware
//
#include <hip/hip_runtime.h>

// ---------------------------------------------------------------------------
// CDNA5 (gfx1250) NAT-1D forward pass.  All heavy GEMMs use
// v_wmma_f32_16x16x32_f16 (wave32, f32 accumulate).  Each wave computes a
// 64x16 output strip: one B fragment per K-step feeds 4 WMMAs (4x B reuse).
// ---------------------------------------------------------------------------

typedef __attribute__((ext_vector_type(16))) _Float16 v16h;
typedef __attribute__((ext_vector_type(8)))  float    v8f;

__device__ __forceinline__ float wave_sum(float v) {
#pragma unroll
  for (int off = 16; off > 0; off >>= 1) v += __shfl_xor(v, off, 32);
  return v;
}

// ---------------------------------------------------------------------------
// GEMM: Y[M,N] = act(A[M,K] @ W[K,N] + bias) + resid
// One wave per 64x16 output strip.  M%64==0, K%32==0, N%16==0.
// act: 0 = none, 1 = exact GELU (applied before residual).
// ---------------------------------------------------------------------------
__global__ void nat_gemm_wmma(const float* __restrict__ A,
                              const float* __restrict__ W,
                              const float* __restrict__ bias,
                              const float* __restrict__ resid,
                              float* __restrict__ Y,
                              int M, int K, int N, int act) {
  int wave = blockIdx.x * (blockDim.x >> 5) + (threadIdx.x >> 5);
  int lane = threadIdx.x & 31;
  int tilesN = N >> 4;
  int strips = (M >> 6) * tilesN;
  if (wave >= strips) return;               // wave-uniform: EXEC stays all-1s
  int tm = (wave / tilesN) << 6;
  int tn = (wave % tilesN) << 4;
  int hf = lane >> 4;                        // 0: lanes 0-15, 1: lanes 16-31
  int lr = lane & 15;

  const float* a0 = A + (size_t)(tm + lr) * K;
  const size_t arstride = (size_t)16 * K;    // 16 rows between sub-tiles
  const float* bcol = W + (tn + lr);

  v8f acc[4] = {{}, {}, {}, {}};
  for (int k0 = 0; k0 < K; k0 += 32) {
    // B fragment (32x16 f16): lane<16 -> K k0+0..15, lane>=16 -> K k0+16..31
    v16h b;
    int kb = k0 + hf * 16;
#pragma unroll
    for (int j = 0; j < 16; ++j)
      b[j] = (_Float16)bcol[(size_t)(kb + j) * N];

    // A fragments (16x32 f16): lane<16 -> K k0+0..15;
    //                           lane>=16 -> K k0+16..31
    int ka0 = k0 + hf * 16;
    int ka1 = ka0 + 8;
#pragma unroll
    for (int t = 0; t < 4; ++t) {
      const float* ar = a0 + (size_t)t * arstride;
      float4 f0 = *(const float4*)(ar + ka0);
      float4 f1 = *(const float4*)(ar + ka0 + 4);
      float4 f2 = *(const float4*)(ar + ka1);
      float4 f3 = *(const float4*)(ar + ka1 + 4);
      v16h a;
      a[0]  = (_Float16)f0.x; a[1]  = (_Float16)f0.y;
      a[2]  = (_Float16)f0.z; a[3]  = (_Float16)f0.w;
      a[4]  = (_Float16)f1.x; a[5]  = (_Float16)f1.y;
      a[6]  = (_Float16)f1.z; a[7]  = (_Float16)f1.w;
      a[8]  = (_Float16)f2.x; a[9]  = (_Float16)f2.y;
      a[10] = (_Float16)f2.z; a[11] = (_Float16)f2.w;
      a[12] = (_Float16)f3.x; a[13] = (_Float16)f3.y;
      a[14] = (_Float16)f3.z; a[15] = (_Float16)f3.w;
      acc[t] = __builtin_amdgcn_wmma_f32_16x16x32_f16(
          false, a, false, b, (short)0, acc[t], false, false);
    }
  }

  // C/D layout: lane<16 -> col=lane, rows 0..7; lane>=16 -> col=lane-16, rows 8..15
  int col = tn + lr;
  float bv = bias ? bias[col] : 0.f;
#pragma unroll
  for (int t = 0; t < 4; ++t) {
#pragma unroll
    for (int r = 0; r < 8; ++r) {
      int row = tm + t * 16 + hf * 8 + r;
      float v = acc[t][r] + bv;
      if (act == 1) v = 0.5f * v * (1.f + erff(v * 0.70710678118654752f));
      if (resid) v += resid[(size_t)row * N + col];
      Y[(size_t)row * N + col] = v;
    }
  }
}

// ---------------------------------------------------------------------------
// LayerNorm over last dim C (C % 32 == 0).  One wave per row.  In-place safe.
// ---------------------------------------------------------------------------
__global__ void nat_ln(const float* __restrict__ X,
                       const float* __restrict__ g,
                       const float* __restrict__ bta,
                       float* __restrict__ Y, int rows, int C) {
  int wave = blockIdx.x * (blockDim.x >> 5) + (threadIdx.x >> 5);
  int lane = threadIdx.x & 31;
  if (wave >= rows) return;
  const float* x = X + (size_t)wave * C;
  float* y = Y + (size_t)wave * C;
  float s = 0.f, s2 = 0.f;
  for (int i = lane; i < C; i += 32) { float v = x[i]; s += v; s2 += v * v; }
  s = wave_sum(s); s2 = wave_sum(s2);
  float mu = s / C;
  float var = s2 / C - mu * mu;
  float inv = rsqrtf(var + 1e-5f);
  for (int i = lane; i < C; i += 32) y[i] = (x[i] - mu) * inv * g[i] + bta[i];
}

// ---------------------------------------------------------------------------
// Neighborhood attention, window 7, head_dim == 32 == wave width.
// One wave per (b, n, head); lane == d.  Out-of-range window rows use qkv_b
// (they come from the zero-padded input in the reference).  Output written in
// the reference's scrambled (B, h, d, n) -> (B, N, C) layout.
// ---------------------------------------------------------------------------
__global__ void nat_attn_k(const float* __restrict__ qkv,
                           const float* __restrict__ qkv_b,
                           const float* __restrict__ pos,
                           float* __restrict__ out,
                           int B, int N, int C, int heads) {
  int wave = blockIdx.x * (blockDim.x >> 5) + (threadIdx.x >> 5);
  int lane = threadIdx.x & 31;
  int total = B * N * heads;
  if (wave >= total) return;
  int h = wave % heads;
  int n = (wave / heads) % N;
  int b = wave / (heads * N);
  const float scale = 0.17677669529663687f;   // 32^-0.5
  int colq = h * 32 + lane;

  float qd[7], kd[7], vd[7];
#pragma unroll
  for (int j = 0; j < 7; ++j) {
    int r = n + j - 3;
    float q, k, v;
    if (r >= 0 && r < N) {
      const float* row = qkv + (size_t)(b * N + r) * (3 * C);
      q = row[colq]; k = row[C + colq]; v = row[2 * C + colq];
    } else {
      q = qkv_b[colq]; k = qkv_b[C + colq]; v = qkv_b[2 * C + colq];
    }
    qd[j] = q * scale; kd[j] = k; vd[j] = v;
  }

  float outd = 0.f;
#pragma unroll
  for (int qi = 0; qi < 7; ++qi) {
    float l[7];
#pragma unroll
    for (int ki = 0; ki < 7; ++ki)
      l[ki] = wave_sum(qd[qi] * kd[ki]) + pos[qi * 7 + ki];
    float m = l[0];
#pragma unroll
    for (int ki = 1; ki < 7; ++ki) m = fmaxf(m, l[ki]);
    float ssum = 0.f, e[7];
#pragma unroll
    for (int ki = 0; ki < 7; ++ki) { e[ki] = expf(l[ki] - m); ssum += e[ki]; }
    float inv = 1.f / ssum;
#pragma unroll
    for (int ki = 0; ki < 7; ++ki) outd += e[ki] * inv * vd[ki];
  }
  out[(size_t)b * C * N + (size_t)(h * 32 + lane) * N + n] = outd;
}

// ---------------------------------------------------------------------------
// Conv1d k=3, stride 2, pad 1.  w layout (Cout, Cin, 3).
// in_nlc / out_nlc select (B,L,C) vs (B,C,L) layouts.
// ---------------------------------------------------------------------------
__global__ void nat_conv(const float* __restrict__ x,
                         const float* __restrict__ w,
                         const float* __restrict__ bias,
                         float* __restrict__ y,
                         int B, int Cin, int Lin, int Cout, int Lout,
                         int in_nlc, int out_nlc) {
  int idx = blockIdx.x * blockDim.x + threadIdx.x;
  int total = B * Cout * Lout;
  if (idx >= total) return;
  int lo = idx % Lout;
  int co = (idx / Lout) % Cout;
  int b = idx / (Lout * Cout);
  float acc = bias ? bias[co] : 0.f;
  for (int kk = 0; kk < 3; ++kk) {
    int li = 2 * lo + kk - 1;
    if (li < 0 || li >= Lin) continue;
    const float* wp = w + ((size_t)co * Cin) * 3 + kk;
    if (in_nlc) {
      const float* xp = x + ((size_t)b * Lin + li) * Cin;
      for (int ci = 0; ci < Cin; ++ci) acc += wp[ci * 3] * xp[ci];
    } else {
      const float* xp = x + ((size_t)b * Cin) * Lin + li;
      for (int ci = 0; ci < Cin; ++ci) acc += wp[ci * 3] * xp[(size_t)ci * Lin];
    }
  }
  if (out_nlc) y[((size_t)b * Lout + lo) * Cout + co] = acc;
  else         y[idx] = acc;
}

// ---------------------------------------------------------------------------
// Head: mean over N, classifier, softmax.  One block per batch element.
// ---------------------------------------------------------------------------
__global__ void nat_head(const float* __restrict__ hin,
                         const float* __restrict__ cls_w,
                         const float* __restrict__ cls_b,
                         float* __restrict__ out, int N, int C, int NC) {
  int b = blockIdx.x;
  __shared__ float mean[512];
  __shared__ float logits[16];
  for (int c = threadIdx.x; c < C; c += blockDim.x) {
    float s = 0.f;
    for (int n = 0; n < N; ++n) s += hin[((size_t)b * N + n) * C + c];
    mean[c] = s / (float)N;
  }
  __syncthreads();
  for (int j = threadIdx.x; j < NC; j += blockDim.x) {
    float s = cls_b[j];
    for (int c = 0; c < C; ++c) s += mean[c] * cls_w[c * NC + j];
    logits[j] = s;
  }
  __syncthreads();
  if (threadIdx.x == 0) {
    float m = logits[0];
    for (int j = 1; j < NC; ++j) m = fmaxf(m, logits[j]);
    float ssum = 0.f;
    for (int j = 0; j < NC; ++j) { float e = expf(logits[j] - m); logits[j] = e; ssum += e; }
    float inv = 1.f / ssum;
    for (int j = 0; j < NC; ++j) out[b * NC + j] = logits[j] * inv;
  }
}

// ---------------------------------------------------------------------------
// Host orchestration
// ---------------------------------------------------------------------------
struct LayerP {
  const float *fc1_b, *fc1_w, *fc2_b, *fc2_w;
  const float *ln1_b, *ln1_g, *ln2_b, *ln2_g;
  const float *pos, *proj_b, *proj_w, *qkv_b, *qkv_w;
};
struct DownP { const float *ln_b, *ln_g, *w; };

extern "C" void kernel_launch(void* const* d_in, const int* in_sizes, int n_in,
                              void* d_out, int out_size, void* d_ws, size_t ws_size,
                              hipStream_t stream) {
  (void)in_sizes; (void)n_in; (void)out_size; (void)ws_size;
  // ---- unpack params (JAX pytree order: dict keys sorted recursively) ----
  int cur = 0;
  auto nx = [&]() { return (const float*)d_in[cur++]; };
  const float* cls_b  = nx();
  const float* cls_w  = nx();
  const float* norm_b = nx();
  const float* norm_g = nx();
  const int depths[4] = {2, 2, 6, 2};
  DownP downs[3];
  LayerP layers[12];
  int li = 0;
  for (int s = 0; s < 4; ++s) {
    if (s < 3) { downs[s].ln_b = nx(); downs[s].ln_g = nx(); downs[s].w = nx(); }
    for (int l = 0; l < depths[s]; ++l) {
      LayerP& L = layers[li++];
      L.fc1_b = nx(); L.fc1_w = nx(); L.fc2_b = nx(); L.fc2_w = nx();
      L.ln1_b = nx(); L.ln1_g = nx(); L.ln2_b = nx(); L.ln2_g = nx();
      L.pos = nx(); L.proj_b = nx(); L.proj_w = nx(); L.qkv_b = nx(); L.qkv_w = nx();
    }
  }
  const float* tok_b1  = nx();
  const float* tok_b2  = nx();
  const float* tok_lnb = nx();
  const float* tok_lng = nx();
  const float* tok_w1  = nx();
  const float* tok_w2  = nx();
  const float* x       = nx();

  // ---- workspace carve (floats). B*N*C == 4,194,304 at every stage ----
  const int B = 64;
  const size_t HSZ = 4194304;           // B*N*C
  float* p   = (float*)d_ws;
  float* hA  = p; p += HSZ;
  float* hB  = p; p += HSZ;
  float* lnb = p; p += HSZ;             // LN output / GEMM input
  float* atb = p; p += HSZ;             // attention output
  float* sb  = p; p += 4 * HSZ;         // shared: qkv (3*HSZ) / mlp hidden (4*HSZ)

  auto gemm = [&](const float* A, const float* W, const float* bias,
                  const float* resid, float* Y, int M, int K, int N, int act) {
    int waves = (M >> 6) * (N >> 4);    // 64x16 strip per wave
    nat_gemm_wmma<<<(waves + 7) / 8, 256, 0, stream>>>(A, W, bias, resid, Y, M, K, N, act);
  };
  auto ln = [&](const float* X, const float* g, const float* bta, float* Y,
                int rows, int C) {
    nat_ln<<<(rows + 7) / 8, 256, 0, stream>>>(X, g, bta, Y, rows, C);
  };

  // ---- tokenizer: conv(3->32, 4096->2048), conv(32->64, 2048->1024), LN ----
  {
    int tot1 = B * 32 * 2048;
    nat_conv<<<(tot1 + 255) / 256, 256, 0, stream>>>(x, tok_w1, tok_b1, sb,
                                                     B, 3, 4096, 32, 2048, 0, 0);
    int tot2 = B * 64 * 1024;
    nat_conv<<<(tot2 + 255) / 256, 256, 0, stream>>>(sb, tok_w2, tok_b2, hA,
                                                     B, 32, 2048, 64, 1024, 0, 1);
    ln(hA, tok_lng, tok_lnb, hA, B * 1024, 64);
  }

  // ---- stages ----
  float* h = hA;
  float* hAlt = hB;
  const int dims[4]  = {64, 128, 256, 512};
  const int Ns[4]    = {1024, 512, 256, 128};
  const int heads[4] = {2, 4, 8, 16};
  li = 0;
  for (int s = 0; s < 4; ++s) {
    int C = dims[s], N = Ns[s], M = B * N, nh = heads[s];
    for (int l = 0; l < depths[s]; ++l) {
      LayerP& L = layers[li++];
      // attention block
      ln(h, L.ln1_g, L.ln1_b, lnb, M, C);
      gemm(lnb, L.qkv_w, L.qkv_b, nullptr, sb, M, C, 3 * C, 0);
      {
        int waves = B * N * nh;
        nat_attn_k<<<(waves + 7) / 8, 256, 0, stream>>>(sb, L.qkv_b, L.pos, atb,
                                                        B, N, C, nh);
      }
      gemm(atb, L.proj_w, L.proj_b, h, h, M, C, C, 0);
      // mlp block
      ln(h, L.ln2_g, L.ln2_b, lnb, M, C);
      gemm(lnb, L.fc1_w, L.fc1_b, nullptr, sb, M, C, 4 * C, 1);
      gemm(sb, L.fc2_w, L.fc2_b, h, h, M, 4 * C, C, 0);
    }
    if (s < 3) {
      int tot = B * (2 * C) * (N / 2);
      nat_conv<<<(tot + 255) / 256, 256, 0, stream>>>(h, downs[s].w, nullptr, hAlt,
                                                      B, C, N, 2 * C, N / 2, 1, 1);
      ln(hAlt, downs[s].ln_g, downs[s].ln_b, hAlt, B * (N / 2), 2 * C);
      float* t = h; h = hAlt; hAlt = t;
    }
  }

  // ---- head: LN, mean over N=128, classifier 512->10, softmax ----
  ln(h, norm_g, norm_b, lnb, B * 128, 512);
  nat_head<<<B, 256, 0, stream>>>(lnb, cls_w, cls_b, (float*)d_out, 128, 512, 10);
}
